// PointTransformerLayer_34308198761275
// MI455X (gfx1250) — matrix operand order repaired and checked by reading, not compiled, not run on hardware
//
#include <hip/hip_runtime.h>

#define BB 4
#define NN 8192
#define DD 128
#define KK 16
#define BN (BB*NN)

typedef __bf16 bf16;
typedef __attribute__((ext_vector_type(16))) __bf16 v16bf;
typedef __attribute__((ext_vector_type(8)))  __bf16 v8bf;
typedef __attribute__((ext_vector_type(4)))  __bf16 v4bf;
typedef __attribute__((ext_vector_type(8)))  float  v8f;

// Async global->LDS path (CDNA5 GLOBAL_LOAD_ASYNC_TO_LDS_B128, ASYNCcnt).
// Guarded: falls back to the VGPR copy loop if the builtins are absent.
#if defined(__has_builtin)
#if __has_builtin(__builtin_amdgcn_global_load_async_to_lds_b128) && \
    __has_builtin(__builtin_amdgcn_s_wait_asynccnt)
#define USE_ASYNC_LDS 1
#endif
#endif

static __device__ __forceinline__ v8f wmma_bf16(v16bf a, v16bf b, v8f c) {
  // D = A(16x32 bf16) * B(32x16 bf16) + C(16x16 f32)
  return __builtin_amdgcn_wmma_f32_16x16x32_bf16(false, a, false, b, (short)0, c, false, false);
}

static __device__ __forceinline__ v16bf cat8(v8bf lo, v8bf hi) {
  return __builtin_shufflevector(lo, hi, 0,1,2,3,4,5,6,7,8,9,10,11,12,13,14,15);
}

// Stage a 32KB (128x128 bf16) matrix into LDS with 256 threads.
static __device__ __forceinline__ void stage_lds_32kb(const bf16* __restrict__ src,
                                                      bf16* dst_lds, int tid) {
#ifdef USE_ASYNC_LDS
  // Builtin signature (from hipcc diagnostic): arg0 = AS1 int4*, 16B granules.
  typedef int v4i __attribute__((vector_size(16)));
  __attribute__((address_space(1))) v4i* g =
      (__attribute__((address_space(1))) v4i*)(char*)src;
  __attribute__((address_space(3))) v4i* l =
      (__attribute__((address_space(3))) v4i*)dst_lds;
  for (int i = tid; i < DD*DD/8; i += 256)
    __builtin_amdgcn_global_load_async_to_lds_b128(g + i, l + i, 0, 0);
  __builtin_amdgcn_s_wait_asynccnt(0);
#else
  for (int i = tid; i < DD*DD/8; i += 256)
    ((uint4*)dst_lds)[i] = ((const uint4*)src)[i];
#endif
}

// ---------------------------------------------------------------------------
// Prep: cq = rowsum(Wq), ck = rowsum(Wk), Bq/Bk = sum of biases,
//       bf16-transposed copies of Wv, Wo, Wp2 ((out,in) layout for WMMA B frags)
// ---------------------------------------------------------------------------
__global__ __launch_bounds__(256) void prep_kernel(
    const float* __restrict__ Wq, const float* __restrict__ bq,
    const float* __restrict__ Wk, const float* __restrict__ bk,
    const float* __restrict__ Wv, const float* __restrict__ Wo,
    const float* __restrict__ Wp2,
    float* __restrict__ cq, float* __restrict__ ck, float* __restrict__ Bqk,
    bf16* __restrict__ Wvt, bf16* __restrict__ Wot, bf16* __restrict__ Wp2t) {
  int t = threadIdx.x;
  if (t < DD) {
    float s0 = 0.f, s1 = 0.f;
    for (int j = 0; j < DD; ++j) { s0 += Wq[t*DD + j]; s1 += Wk[t*DD + j]; }
    cq[t] = s0; ck[t] = s1;
  }
  if (t == 0) {
    float a = 0.f, b = 0.f;
    for (int j = 0; j < DD; ++j) { a += bq[j]; b += bk[j]; }
    Bqk[0] = a; Bqk[1] = b;
  }
  for (int e = t; e < DD*DD; e += 256) {
    int n = e >> 7, k = e & (DD-1);        // Wt[n][k] = W[k][n]
    Wvt[e]  = (bf16)Wv[k*DD + n];
    Wot[e]  = (bf16)Wo[k*DD + n];
    Wp2t[e] = (bf16)Wp2[k*DD + n];
  }
}

// ---------------------------------------------------------------------------
// featprep: feat -> bf16 copy, plus per-point scalars
//   sq[p] = feat[p] . cq + Bq     (== sum_d of q row)
//   sk[p] = feat[p] . ck + Bk     (== sum_d of k row)
// One wave per row; 8 waves per block.
// ---------------------------------------------------------------------------
__global__ __launch_bounds__(256) void featprep_kernel(
    const float* __restrict__ feat, const float* __restrict__ cq,
    const float* __restrict__ ck, const float* __restrict__ Bqk,
    bf16* __restrict__ featb, float* __restrict__ sq, float* __restrict__ sk) {
  int w = threadIdx.x >> 5, l = threadIdx.x & 31;
  int row = blockIdx.x * 8 + w;
  const float4 f  = *(const float4*)(feat + (size_t)row*DD + l*4);
  const float4 c0 = *(const float4*)(cq + l*4);
  const float4 c1 = *(const float4*)(ck + l*4);
  v4bf o; o[0] = (bf16)f.x; o[1] = (bf16)f.y; o[2] = (bf16)f.z; o[3] = (bf16)f.w;
  *(v4bf*)(featb + (size_t)row*DD + l*4) = o;
  float dq = f.x*c0.x + f.y*c0.y + f.z*c0.z + f.w*c0.w;
  float dk = f.x*c1.x + f.y*c1.y + f.z*c1.z + f.w*c1.w;
  #pragma unroll
  for (int m = 16; m >= 1; m >>= 1) {
    dq += __shfl_xor(dq, m, 32);
    dk += __shfl_xor(dk, m, 32);
  }
  if (l == 0) { sq[row] = dq + Bqk[0]; sk[row] = dk + Bqk[1]; }
}

// ---------------------------------------------------------------------------
// KNN: one thread per query, candidates streamed through LDS in 512-pt tiles,
// sorted 16-entry register list with guarded insertion.
// ---------------------------------------------------------------------------
__global__ __launch_bounds__(256) void knn_kernel(
    const float* __restrict__ xyz, int* __restrict__ idxb) {
  __shared__ float cx[512], cy[512], cz[512];
  int tid = threadIdx.x;
  int g = blockIdx.x * 256 + tid;        // global point id (blocks never straddle batches)
  int b = g / NN;
  float qx = xyz[(size_t)g*3 + 0];
  float qy = xyz[(size_t)g*3 + 1];
  float qz = xyz[(size_t)g*3 + 2];
  float bd[KK]; int bi[KK];
  #pragma unroll
  for (int t = 0; t < KK; ++t) { bd[t] = 3.4e38f; bi[t] = 0; }
  int base = b * NN;
  for (int tile = 0; tile < NN; tile += 512) {
    for (int j = tid; j < 512; j += 256) {
      int p = base + tile + j;
      cx[j] = xyz[(size_t)p*3 + 0];
      cy[j] = xyz[(size_t)p*3 + 1];
      cz[j] = xyz[(size_t)p*3 + 2];
    }
    __syncthreads();
    for (int j = 0; j < 512; ++j) {
      float dx = qx - cx[j], dy = qy - cy[j], dz = qz - cz[j];
      float d = fmaf(dx, dx, fmaf(dy, dy, dz*dz));
      if (d < bd[KK-1]) {                 // rarely taken after warm-up
        float cd = d; int ci = tile + j;  // within-batch index, matches reference
        #pragma unroll
        for (int t = 0; t < KK; ++t) {
          bool s = cd < bd[t];
          float td = bd[t]; int ti = bi[t];
          if (s) { bd[t] = cd; bi[t] = ci; cd = td; ci = ti; }
        }
      }
    }
    __syncthreads();
  }
  #pragma unroll
  for (int t = 0; t < KK; ++t) idxb[(size_t)g*KK + t] = bi[t];
}

// ---------------------------------------------------------------------------
// 32768x128 @ 128x128 bf16 WMMA GEMM. Bt is (out,in) bf16, staged in LDS.
// Block = 8 waves, each wave owns 16 rows x all 128 cols (8 C tiles).
// MODE 0: out = A@B + bias            (V projection)
// MODE 1: out = feat + gamma*(A@B+b)  (output projection + residual)
// ---------------------------------------------------------------------------
template<int MODE>
__global__ __launch_bounds__(256) void gemm128_kernel(
    const bf16* __restrict__ A, const bf16* __restrict__ Bt,
    const float* __restrict__ bias, const float* __restrict__ feat,
    const float* __restrict__ gammaPtr, float* __restrict__ out) {
  __shared__ bf16 bs[DD*DD];
  int tid = threadIdx.x;
  stage_lds_32kb(Bt, bs, tid);
  __syncthreads();

  int w = tid >> 5, l = tid & 31;
  int rbase = blockIdx.x * 128 + w * 16;
  int hi = l >> 4;                 // lane half
  int mrow = rbase + (l & 15);     // A row for this lane
  int ncol = l & 15;               // local col inside C tile
  v8f acc[8];
  #pragma unroll
  for (int nt = 0; nt < 8; ++nt) acc[nt] = (v8f){0,0,0,0,0,0,0,0};

  #pragma unroll
  for (int kt = 0; kt < 4; ++kt) {
    // A fragment: lane holds row (l&15); K = kb..kb+7 (elems 0-7), kb+16..kb+23 (elems 8-15)
    int kb = kt*32 + hi*8;
    v8bf alo = *(const v8bf*)(A + (size_t)mrow*DD + kb);
    v8bf ahi = *(const v8bf*)(A + (size_t)mrow*DD + kb + 16);
    v16bf a = cat8(alo, ahi);
    #pragma unroll
    for (int nt = 0; nt < 8; ++nt) {
      // B fragment: lane holds col n; K = kb..kb+7 (elems 0-7), kb+16..kb+23 (elems 8-15)
      const bf16* bp = bs + (size_t)(nt*16 + ncol)*DD + kt*32 + hi*8;
      v16bf bb = cat8(*(const v8bf*)bp, *(const v8bf*)(bp + 16));
      acc[nt] = wmma_bf16(a, bb, acc[nt]);
    }
  }

  float gamma = (MODE == 1) ? *gammaPtr : 0.f;
  #pragma unroll
  for (int nt = 0; nt < 8; ++nt) {
    int col = nt*16 + ncol;
    float bcol = bias[col];
    #pragma unroll
    for (int r = 0; r < 8; ++r) {
      int rr = rbase + r + hi*8;   // C layout: lanes 0-15 rows 0-7, lanes 16-31 rows 8-15
      float val = acc[nt][r] + bcol;
      if (MODE == 0) out[(size_t)rr*DD + col] = val;
      else           out[(size_t)rr*DD + col] =
                        feat[(size_t)rr*DD + col] + gamma * val;
    }
  }
}

// ---------------------------------------------------------------------------
// Fused pos-enc + attention + aggregation. One wave per query point.
// pos(16x128) = relu(rel_pos@Wp1+bp1) @ Wp2 + bp2 via 32 WMMAs, kept in accs.
// score/softmax/agg done with cross-lane shuffles on the accumulator layout.
// ---------------------------------------------------------------------------
__global__ __launch_bounds__(256) void attn_kernel(
    const float* __restrict__ xyz, const float* __restrict__ Wp1,
    const float* __restrict__ bp1, const float* __restrict__ bp2,
    const bf16* __restrict__ Wp2t, const int* __restrict__ idxb,
    const float* __restrict__ vbuf, const float* __restrict__ sq,
    const float* __restrict__ sk, bf16* __restrict__ aggb) {
  __shared__ bf16  wp2s[DD*DD];
  __shared__ float wp1s[3*DD];
  __shared__ float bp1s[DD];
  __shared__ float bp2s[DD];
  int tid = threadIdx.x;
  stage_lds_32kb(Wp2t, wp2s, tid);
  for (int i = tid; i < 3*DD; i += 256) wp1s[i] = Wp1[i];
  for (int i = tid; i < DD; i += 256) { bp1s[i] = bp1[i]; bp2s[i] = bp2[i]; }
  __syncthreads();

  int w = tid >> 5, l = tid & 31;
  int q = blockIdx.x * 8 + w;        // global point id
  int b = q / NN;
  int hi = l >> 4;
  int m  = l & 15;

  // lane holds neighbor id for k = (l&15)
  int idxval = idxb[(size_t)q*KK + m];
  int nidk[8];
  #pragma unroll
  for (int r = 0; r < 8; ++r) {
    nidk[r] = __shfl(idxval, r + hi*8, 32);   // k = r (lo half) / r+8 (hi half)
    __builtin_prefetch(vbuf + ((size_t)b*NN + nidk[r])*DD + m, 0, 0);
  }

  // rel_pos for this lane's neighbor (A-matrix row m)
  float qx = xyz[(size_t)q*3+0], qy = xyz[(size_t)q*3+1], qz = xyz[(size_t)q*3+2];
  size_t nb = ((size_t)b*NN + idxval)*3;
  float px = qx - xyz[nb], py = qy - xyz[nb+1], pz = qz - xyz[nb+2];

  // h = relu(rel_pos @ Wp1 + bp1), packed directly into WMMA A fragments
  v16bf af[4];
  #pragma unroll
  for (int kt = 0; kt < 4; ++kt) {
    int cb = kt*32 + hi*8;
    #pragma unroll
    for (int j = 0; j < 8; ++j) {
      int c1 = cb + j;
      float h1 = fmaxf(0.f, fmaf(px, wp1s[c1], fmaf(py, wp1s[DD+c1],
                         fmaf(pz, wp1s[2*DD+c1], bp1s[c1]))));
      af[kt][j] = (bf16)h1;
      int c2 = cb + 16 + j;
      float h2 = fmaxf(0.f, fmaf(px, wp1s[c2], fmaf(py, wp1s[DD+c2],
                         fmaf(pz, wp1s[2*DD+c2], bp1s[c2]))));
      af[kt][8+j] = (bf16)h2;
    }
  }

  // pos = h @ Wp2 : rows = neighbors (16), cols = channels (128)
  v8f acc[8];
  #pragma unroll
  for (int nt = 0; nt < 8; ++nt) acc[nt] = (v8f){0,0,0,0,0,0,0,0};
  #pragma unroll
  for (int kt = 0; kt < 4; ++kt) {
    #pragma unroll
    for (int nt = 0; nt < 8; ++nt) {
      const bf16* bp = wp2s + (size_t)(nt*16 + m)*DD + kt*32 + hi*8;
      v16bf bb = cat8(*(const v8bf*)bp, *(const v8bf*)(bp + 16));
      acc[nt] = wmma_bf16(af[kt], bb, acc[nt]);
    }
  }
  #pragma unroll
  for (int nt = 0; nt < 8; ++nt) {
    float bc = bp2s[nt*16 + m];
    #pragma unroll
    for (int r = 0; r < 8; ++r) acc[nt][r] += bc;
  }

  // score[k] = sq[q] - sk[nbr_k] + sum_d pos[k][d]   (row-sum via xor-reduce over 16 lanes)
  float score[8];
  float sqn = sq[q];
  #pragma unroll
  for (int r = 0; r < 8; ++r) {
    float s = 0.f;
    #pragma unroll
    for (int nt = 0; nt < 8; ++nt) s += acc[nt][r];
    #pragma unroll
    for (int mk = 8; mk >= 1; mk >>= 1) s += __shfl_xor(s, mk, 32);
    score[r] = sqn - sk[(size_t)b*NN + nidk[r]] + s;
  }

  // softmax over 16 neighbors (8 per half, combine halves with lane-16 xor)
  float mx = -3.4e38f;
  #pragma unroll
  for (int r = 0; r < 8; ++r) mx = fmaxf(mx, score[r]);
  mx = fmaxf(mx, __shfl_xor(mx, 16, 32));
  float e[8], es = 0.f;
  #pragma unroll
  for (int r = 0; r < 8; ++r) { e[r] = __expf(score[r] - mx); es += e[r]; }
  es += __shfl_xor(es, 16, 32);
  float inv = 1.f / es;

  // agg_d = sum_k attn_k * (v[k][d] + pos[k][d]); halves combined with lane-16 xor
  #pragma unroll
  for (int nt = 0; nt < 8; ++nt) {
    int col = nt*16 + m;
    float p = 0.f;
    #pragma unroll
    for (int r = 0; r < 8; ++r) {
      float vv = vbuf[((size_t)b*NN + nidk[r])*DD + col];
      p = fmaf(e[r]*inv, acc[nt][r] + vv, p);
    }
    p += __shfl_xor(p, 16, 32);
    if (hi == 0) aggb[(size_t)q*DD + col] = (bf16)p;
  }
}

// ---------------------------------------------------------------------------
extern "C" void kernel_launch(void* const* d_in, const int* in_sizes, int n_in,
                              void* d_out, int out_size, void* d_ws, size_t ws_size,
                              hipStream_t stream) {
  const float* xyz  = (const float*)d_in[0];
  const float* feat = (const float*)d_in[1];
  const float* Wq   = (const float*)d_in[2];
  const float* bq   = (const float*)d_in[3];
  const float* Wk   = (const float*)d_in[4];
  const float* bk   = (const float*)d_in[5];
  const float* Wv   = (const float*)d_in[6];
  const float* bv   = (const float*)d_in[7];
  const float* Wp1  = (const float*)d_in[8];
  const float* bp1  = (const float*)d_in[9];
  const float* Wp2  = (const float*)d_in[10];
  const float* bp2  = (const float*)d_in[11];
  const float* Wo   = (const float*)d_in[12];
  const float* bo   = (const float*)d_in[13];
  const float* gamma= (const float*)d_in[14];
  (void)in_sizes; (void)n_in; (void)out_size; (void)ws_size;

  char* ws = (char*)d_ws;
  size_t off = 0;
  int*   idxb  = (int*)(ws + off);   off += (size_t)BN*KK*sizeof(int);   // 2 MB
  float* vbuf  = (float*)(ws + off); off += (size_t)BN*DD*sizeof(float); // 16 MB
  bf16*  featb = (bf16*)(ws + off);  off += (size_t)BN*DD*sizeof(bf16);  // 8 MB
  bf16*  aggb  = (bf16*)(ws + off);  off += (size_t)BN*DD*sizeof(bf16);  // 8 MB
  float* sqb   = (float*)(ws + off); off += (size_t)BN*sizeof(float);
  float* skb   = (float*)(ws + off); off += (size_t)BN*sizeof(float);
  float* cq    = (float*)(ws + off); off += 512;
  float* ck    = (float*)(ws + off); off += 512;
  float* Bqk   = (float*)(ws + off); off += 256;
  bf16*  Wvt   = (bf16*)(ws + off);  off += (size_t)DD*DD*sizeof(bf16);
  bf16*  Wot   = (bf16*)(ws + off);  off += (size_t)DD*DD*sizeof(bf16);
  bf16*  Wp2t  = (bf16*)(ws + off);  off += (size_t)DD*DD*sizeof(bf16);

  prep_kernel<<<1, 256, 0, stream>>>(Wq, bq, Wk, bk, Wv, Wo, Wp2,
                                     cq, ck, Bqk, Wvt, Wot, Wp2t);
  featprep_kernel<<<BN/8, 256, 0, stream>>>(feat, cq, ck, Bqk, featb, sqb, skb);
  knn_kernel<<<BN/256, 256, 0, stream>>>(xyz, idxb);
  gemm128_kernel<0><<<BN/128, 256, 0, stream>>>(featb, Wvt, bv, nullptr, nullptr, vbuf);
  attn_kernel<<<BN/8, 256, 0, stream>>>(xyz, Wp1, bp1, bp2, Wp2t, idxb,
                                        vbuf, sqb, skb, aggb);
  gemm128_kernel<1><<<BN/128, 256, 0, stream>>>(aggb, Wot, bo, feat, gamma,
                                                (float*)d_out);
}
